// QConv2d_79963701117259
// MI455X (gfx1250) — 8-bit matrix operand order repaired and checked by reading, compiled, not run on hardware
//
#include <hip/hip_runtime.h>

typedef int v8i __attribute__((ext_vector_type(8)));

#define HW    112
#define PHW   114
#define CIN   64
#define COUT  128
#define XQP_BYTES ((size_t)32 * PHW * PHW * CIN)

__global__ __launch_bounds__(256) void x_to_nhwc_i8_pad(const float* __restrict__ x,
                                                        signed char* __restrict__ xq) {
    const int hp = blockIdx.x;
    const int n  = blockIdx.y;
    const int t  = threadIdx.x;
    signed char* dst = xq + (((size_t)n * PHW + hp) * PHW) * CIN;

    if (hp == 0 || hp == PHW - 1) {
        int4 z; z.x = z.y = z.z = z.w = 0;
        for (int i = t; i < (PHW * CIN) / 16; i += 256)
            reinterpret_cast<int4*>(dst)[i] = z;
        return;
    }

    __shared__ signed char lds[PHW * CIN];
    if (t < 8) {
        int4 z; z.x = z.y = z.z = z.w = 0;
        int ofs = (t < 4) ? t * 16 : (PHW - 1) * CIN + (t - 4) * 16;
        *reinterpret_cast<int4*>(lds + ofs) = z;
    }

    const int h = hp - 1;
    const int c = t >> 2;
    const int q = t & 3;
    const float* src = x + (((size_t)n * CIN + c) * HW + h) * HW + q * 28;
#pragma unroll
    for (int i = 0; i < 7; ++i) {
        float4 f = reinterpret_cast<const float4*>(src)[i];
        int w = q * 28 + i * 4;
        lds[(w + 1) * CIN + c] = (signed char)(int)f.x;
        lds[(w + 2) * CIN + c] = (signed char)(int)f.y;
        lds[(w + 3) * CIN + c] = (signed char)(int)f.z;
        lds[(w + 4) * CIN + c] = (signed char)(int)f.w;
    }
    __syncthreads();

    for (int ofs = t * 16; ofs < PHW * CIN; ofs += 256 * 16)
        *reinterpret_cast<int4*>(dst + ofs) = *reinterpret_cast<const int4*>(lds + ofs);
}

__global__ __launch_bounds__(256) void pack_weights(const float* __restrict__ w,
                                                    unsigned int* __restrict__ wp) {
    const int d = blockIdx.x * blockDim.x + threadIdx.x;
    if (d >= 8 * 9 * 32 * 8) return;
    const int v    = d & 7;
    const int lane = (d >> 3) & 31;
    const int gt   = d >> 8;
    const int tap  = gt % 9;
    const int g    = gt / 9;
    const int m  = lane & 15;
    const int hi = lane >> 4;
    const int r = tap / 3, s = tap % 3;
    const int ko = g * 16 + m;
    unsigned int pk = 0;
#pragma unroll
    for (int b = 0; b < 4; ++b) {
        int k = (v >> 2) * 32 + hi * 16 + (v & 3) * 4 + b;
        int val = (int)w[(((size_t)ko * CIN + k) * 3 + r) * 3 + s];
        pk |= ((unsigned int)(val & 0xff)) << (8 * b);
    }
    wp[d] = pk;
}

__global__ __launch_bounds__(256) void qconv_wmma(const signed char* __restrict__ xq,
                                                  const uint4* __restrict__ wp,
                                                  const float* __restrict__ bias,
                                                  float* __restrict__ out) {
    const int lane = threadIdx.x & 31;
    const int g    = threadIdx.x >> 5;
    const int w0   = blockIdx.x * 16;
    const int h0   = blockIdx.y * 4;
    const int n    = blockIdx.z;

    const int mofs = (lane >> 4) * 8;
    const float* bb = bias + g * 16 + mofs;
    float4 bf0 = *reinterpret_cast<const float4*>(bb);
    float4 bf1 = *reinterpret_cast<const float4*>(bb + 4);

    v8i a[9];
#pragma unroll
    for (int t = 0; t < 9; ++t) {
        const uint4* p = wp + ((g * 9 + t) * 64 + lane * 2);
        uint4 lo = p[0], hi = p[1];
        a[t][0] = (int)lo.x; a[t][1] = (int)lo.y; a[t][2] = (int)lo.z; a[t][3] = (int)lo.w;
        a[t][4] = (int)hi.x; a[t][5] = (int)hi.y; a[t][6] = (int)hi.z; a[t][7] = (int)hi.w;
    }

    v8i acc[4];
#pragma unroll
    for (int rr = 0; rr < 4; ++rr) {
        acc[rr][0] = (int)bf0.x; acc[rr][1] = (int)bf0.y;
        acc[rr][2] = (int)bf0.z; acc[rr][3] = (int)bf0.w;
        acc[rr][4] = (int)bf1.x; acc[rr][5] = (int)bf1.y;
        acc[rr][6] = (int)bf1.z; acc[rr][7] = (int)bf1.w;
    }

    const int col = lane & 15;
    const int kh  = (lane >> 4) * 16;
    const signed char* xbase =
        xq + (((size_t)n * PHW + h0) * PHW + w0 + col) * CIN + kh;

#pragma unroll
    for (int s = 0; s < 3; ++s) {
#pragma unroll
        for (int hh = 0; hh < 6; ++hh) {
            const signed char* bp = xbase + ((size_t)hh * PHW + s) * CIN;
            uint4 lo = *reinterpret_cast<const uint4*>(bp);
            uint4 hi = *reinterpret_cast<const uint4*>(bp + 32);
            v8i b;
            b[0] = (int)lo.x; b[1] = (int)lo.y; b[2] = (int)lo.z; b[3] = (int)lo.w;
            b[4] = (int)hi.x; b[5] = (int)hi.y; b[6] = (int)hi.z; b[7] = (int)hi.w;
#pragma unroll
            for (int r = 0; r < 3; ++r) {
                const int rr = hh - r;
                if (rr >= 0 && rr < 4)
                    acc[rr] = __builtin_amdgcn_wmma_i32_16x16x64_iu8(
                        true, a[r * 3 + s], true, b, acc[rr], false, false);
            }
        }
    }

#pragma unroll
    for (int rr = 0; rr < 4; ++rr) {
        const int h = h0 + rr;
#pragma unroll
        for (int v = 0; v < 8; ++v) {
            const int ch = g * 16 + mofs + v;
            float f = (float)acc[rr][v] * 0.03125f;
            f = fminf(fmaxf(f, -128.0f), 127.0f);
            f = truncf(f);
            __builtin_nontemporal_store(
                f, &out[(((size_t)n * COUT + ch) * HW + h) * HW + (w0 + col)]);
        }
    }
}

extern "C" void kernel_launch(void* const* d_in, const int* in_sizes, int n_in,
                              void* d_out, int out_size, void* d_ws, size_t ws_size,
                              hipStream_t stream) {
    const float* x    = (const float*)d_in[0];
    const float* wgt  = (const float*)d_in[1];
    const float* bias = (const float*)d_in[2];
    float* out = (float*)d_out;

    signed char*  xq  = (signed char*)d_ws;
    unsigned int* wpk = (unsigned int*)((char*)d_ws + XQP_BYTES);

    x_to_nhwc_i8_pad<<<dim3(PHW, 32), 256, 0, stream>>>(x, xq);
    pack_weights<<<dim3(72), 256, 0, stream>>>(wgt, wpk);
    qconv_wmma<<<dim3(HW / 16, HW / 4, 32), 256, 0, stream>>>(
        xq, (const uint4*)wpk, bias, out);
    (void)in_sizes; (void)n_in; (void)out_size; (void)ws_size;
}
